// DetectiveNN_72834055406051
// MI455X (gfx1250) — matrix operand order repaired and checked by reading, not compiled, not run on hardware
//
#include <hip/hip_runtime.h>
#include <hip/hip_bf16.h>
#include <cstdint>

// Problem constants (match reference)
#define Tn   256
#define Bn   64
#define Dn   1024
#define Hn   512
#define Pn   2
#define PBn  (Pn*Bn)      // 128
#define G4H  (4*Hn)       // 2048
#define H2   (2*Hn)       // 1024

typedef __bf16 v16bf __attribute__((ext_vector_type(16)));
typedef __bf16 v8bf  __attribute__((ext_vector_type(8)));
typedef float  v8f   __attribute__((ext_vector_type(8)));

__device__ __forceinline__ float sigmoidf_(float x) { return 1.0f / (1.0f + __expf(-x)); }

// ---------------------------------------------------------------------------
// Utility kernels
// ---------------------------------------------------------------------------
__global__ void k_zero_u32(uint32_t* __restrict__ p, long long n) {
    long long i = (long long)blockIdx.x * blockDim.x + threadIdx.x;
    long long s = (long long)gridDim.x * blockDim.x;
    for (; i < n; i += s) p[i] = 0u;
}

__global__ void k_f32_to_bf16(const float* __restrict__ in, __bf16* __restrict__ out, long long n) {
    long long i = (long long)blockIdx.x * blockDim.x + threadIdx.x;
    long long s = (long long)gridDim.x * blockDim.x;
    for (; i < n; i += s) out[i] = (__bf16)in[i];
}

// Per-batch speaker id + rank-within-speaker (cumsum of one-hot mask).
__global__ void k_pos(const float* __restrict__ qmask, int* __restrict__ spk, int* __restrict__ pos) {
    int b = threadIdx.x;
    if (b >= Bn) return;
    int cnt0 = 0, cnt1 = 0;
    for (int t = 0; t < Tn; ++t) {
        const float* q = qmask + ((size_t)b * Tn + t) * Pn;
        int s = (q[1] > 0.5f) ? 1 : 0;
        spk[b * Tn + t] = s;
        pos[b * Tn + t] = s ? cnt1++ : cnt0++;
    }
}

// Compact each speaker's utterances to front of its own stream (to bf16).
// xp layout: (T, PB, D); stream index = s*B + b, time index = rank.
__global__ void k_gather(const float* __restrict__ U, const int* __restrict__ spk,
                         const int* __restrict__ pos, __bf16* __restrict__ xp) {
    int tb = blockIdx.x;
    int t = tb / Bn, b = tb % Bn;
    int s = spk[b * Tn + t];
    int tp = pos[b * Tn + t];
    const float* src = U + ((size_t)t * Bn + b) * Dn;
    __bf16* dst = xp + ((size_t)tp * PBn + (size_t)s * Bn + b) * Dn;
    for (int k = threadIdx.x; k < Dn; k += blockDim.x) dst[k] = (__bf16)src[k];
}

// Scatter encoded party streams back to original time positions: out[t,b,2H:4H].
__global__ void k_scatter(const float* __restrict__ E, const int* __restrict__ spk,
                          const int* __restrict__ pos, float* __restrict__ out) {
    int tb = blockIdx.x;
    int t = tb / Bn, b = tb % Bn;
    int s = spk[b * Tn + t];
    int tp = pos[b * Tn + t];
    const float* src = E + ((size_t)tp * PBn + (size_t)s * Bn + b) * H2;
    float* dst = out + ((size_t)t * Bn + b) * (4 * Hn) + H2;
    for (int k = threadIdx.x; k < H2; k += blockDim.x) dst[k] = src[k];
}

// ---------------------------------------------------------------------------
// Input-projection GEMM: Y(M,4H) = X(M,K) @ W(4H,K)^T + bias, bf16 in/out, f32 acc.
// One wave computes a 16x64 output tile (A-fragment reused across 4 WMMAs).
// A lane layout (16-bit A 16x32): lanes 0-15 hold M=l16, K = k0..k0+15;
// lanes 16-31 hold K = k0+16..k0+31 -> two contiguous 8-bf16 loads.
// B lane layout (16-bit B 32x16): lanes 0-15 N=l16 K=k0..k0+15; lanes 16-31 K=+16
//   -> one contiguous 16-bf16 load from row-major W.
// ---------------------------------------------------------------------------
__global__ __launch_bounds__(256) void k_gemm_xg(
    const __bf16* __restrict__ X, const __bf16* __restrict__ W,
    const float* __restrict__ bias, __bf16* __restrict__ Y, int M, int K) {
    const int N = G4H;
    const int wave = threadIdx.x >> 5;
    const int lane = threadIdx.x & 31;
    const int l16 = lane & 15, half = lane >> 4;

    long long tile = (long long)blockIdx.x * 8 + wave;
    const int ngrps = N / 64;                 // 32
    int mtile = (int)(tile / ngrps);
    int ngrp  = (int)(tile % ngrps);
    if (mtile * 16 >= M) return;              // wave-uniform

    const __bf16* xr = X + (size_t)(mtile * 16 + l16) * K + half * 16;
    const __bf16* wr[4];
    int ncol[4];
#pragma unroll
    for (int j = 0; j < 4; ++j) {
        int n = ngrp * 64 + j * 16 + l16;
        ncol[j] = n;
        wr[j] = W + (size_t)n * K + half * 16;
    }

    v8f acc[4] = {};
    for (int k0 = 0; k0 < K; k0 += 32) {
        v8bf alo = *(const v8bf*)(xr + k0);
        v8bf ahi = *(const v8bf*)(xr + k0 + 8);
        v16bf a = __builtin_shufflevector(alo, ahi, 0, 1, 2, 3, 4, 5, 6, 7,
                                          8, 9, 10, 11, 12, 13, 14, 15);
#pragma unroll
        for (int j = 0; j < 4; ++j) {
            v16bf bm = *(const v16bf*)(wr[j] + k0);
            acc[j] = __builtin_amdgcn_wmma_f32_16x16x32_bf16(
                false, a, false, bm, (short)0, acc[j], false, false);
        }
    }

    const int rowbase = mtile * 16 + half * 8;   // C/D: lanes 0-15 M=r, lanes 16-31 M=8+r
#pragma unroll
    for (int j = 0; j < 4; ++j) {
        float bv = bias[ncol[j]];
#pragma unroll
        for (int r = 0; r < 8; ++r)
            Y[(size_t)(rowbase + r) * N + ncol[j]] = (__bf16)(acc[j][r] + bv);
    }
}

// ---------------------------------------------------------------------------
// One LSTM time step, both directions fused (dir = blockIdx.x & 1).
// Block = 16 rows x 64 h-cols -> needs 16x256 gates (i,f,g,o x 64), K=512.
// 8 waves x 2 WMMA tiles each; gates + xg staged in LDS; f32 gate math.
// h double-buffered across steps (h_in read-only, h_out written) to avoid the
// cross-block RAW race on the recurrent input.
// ---------------------------------------------------------------------------
__global__ __launch_bounds__(256) void k_lstm_step(
    const __bf16* __restrict__ xg_f, const __bf16* __restrict__ xg_b,
    const __bf16* __restrict__ Whh,           // (2, 4H, H) bf16
    const __bf16* __restrict__ h_in,          // (2, NB, H) bf16
    __bf16* __restrict__ h_out,               // (2, NB, H) bf16
    float* __restrict__ c_st,                 // (2, NB, H) f32
    __bf16* __restrict__ out_bf,              // (T, NB, 2H) or null
    float* __restrict__ out_f32, long long ot_stride, int or_stride,
    int t, int NB) {
    __shared__ float gl[16][257];

    int bx = blockIdx.x;
    const int dir = bx & 1; bx >>= 1;
    const int mb = bx >> 3;                   // row-tile (16 rows)
    const int hb = bx & 7;                    // 64-wide h column block
    const int t_eff = dir ? (Tn - 1 - t) : t;

    const __bf16* xg = dir ? xg_b : xg_f;
    const __bf16* Wd = Whh + (size_t)dir * G4H * Hn;
    const __bf16* hi = h_in + (size_t)dir * NB * Hn;
    __bf16* ho = h_out + (size_t)dir * NB * Hn;
    float* cd = c_st + (size_t)dir * NB * Hn;

    const int wave = threadIdx.x >> 5;
    const int lane = threadIdx.x & 31;
    const int l16 = lane & 15, half = lane >> 4;

    const __bf16* ar = hi + (size_t)(mb * 16 + l16) * Hn + half * 16;
    const __bf16* wr[2];
    int nglob[2];
#pragma unroll
    for (int j = 0; j < 2; ++j) {
        int cidx = wave * 2 + j;              // 0..15 : which 16-col chunk of 4x64 gates
        int g = cidx >> 2;                    // gate (i,f,g,o)
        int cb = (cidx & 3) * 16;             // within-gate column base
        int n = g * Hn + hb * 64 + cb + l16;  // global gate column in 4H
        nglob[j] = n;
        wr[j] = Wd + (size_t)n * Hn + half * 16;
    }

    v8f acc[2] = {};
    for (int k0 = 0; k0 < Hn; k0 += 32) {
        v8bf alo = *(const v8bf*)(ar + k0);
        v8bf ahi = *(const v8bf*)(ar + k0 + 8);
        v16bf a = __builtin_shufflevector(alo, ahi, 0, 1, 2, 3, 4, 5, 6, 7,
                                          8, 9, 10, 11, 12, 13, 14, 15);
#pragma unroll
        for (int j = 0; j < 2; ++j) {
            v16bf bm = *(const v16bf*)(wr[j] + k0);
            acc[j] = __builtin_amdgcn_wmma_f32_16x16x32_bf16(
                false, a, false, bm, (short)0, acc[j], false, false);
        }
    }

    // gates = recurrent + precomputed input projection (bias already folded in)
    const long long xgrow = (long long)t_eff * NB + mb * 16 + half * 8;
#pragma unroll
    for (int j = 0; j < 2; ++j) {
        int colL = (wave * 2 + j) * 16 + l16; // 0..255 within gate tile
#pragma unroll
        for (int r = 0; r < 8; ++r) {
            float xv = (float)xg[(xgrow + r) * G4H + nglob[j]];
            gl[half * 8 + r][colL] = acc[j][r] + xv;
        }
    }
    __syncthreads();

    const int tid = threadIdx.x;
#pragma unroll
    for (int jj = 0; jj < 4; ++jj) {
        int idx = jj * 256 + tid;             // 0..1023 = 16 rows x 64 cols
        int row = idx >> 6, nl = idx & 63;
        float i_ = sigmoidf_(gl[row][nl]);
        float f_ = sigmoidf_(gl[row][64 + nl]);
        float g_ = tanhf(gl[row][128 + nl]);
        float o_ = sigmoidf_(gl[row][192 + nl]);
        int m = mb * 16 + row;
        int hc = hb * 64 + nl;
        size_t ci = (size_t)m * Hn + hc;
        float cn = f_ * cd[ci] + i_ * g_;
        float hn = o_ * tanhf(cn);
        cd[ci] = cn;
        ho[ci] = (__bf16)hn;
        if (out_bf)
            out_bf[((size_t)t_eff * NB + m) * H2 + (size_t)dir * Hn + hc] = (__bf16)hn;
        if (out_f32)
            out_f32[(long long)t_eff * ot_stride + (long long)m * or_stride + dir * Hn + hc] = hn;
    }
}

// ---------------------------------------------------------------------------
// Host orchestration
// ---------------------------------------------------------------------------
extern "C" void kernel_launch(void* const* d_in, const int* in_sizes, int n_in,
                              void* d_out, int out_size, void* d_ws, size_t ws_size,
                              hipStream_t stream) {
    (void)in_sizes; (void)n_in; (void)out_size; (void)ws_size;
    const float* U     = (const float*)d_in[0];
    const float* qmask = (const float*)d_in[1];
    // d_in[2] = seq_lengths (all T, unused)
    const float* rnn_Wih0  = (const float*)d_in[3];
    const float* rnn_Whh0  = (const float*)d_in[4];
    const float* rnn_b0    = (const float*)d_in[5];
    const float* rnn_Wih1  = (const float*)d_in[6];
    const float* rnn_Whh1  = (const float*)d_in[7];
    const float* rnn_b1    = (const float*)d_in[8];
    const float* rnnp_Wih0 = (const float*)d_in[9];
    const float* rnnp_Whh0 = (const float*)d_in[10];
    const float* rnnp_b0   = (const float*)d_in[11];
    const float* rnnp_Wih1 = (const float*)d_in[12];
    const float* rnnp_Whh1 = (const float*)d_in[13];
    const float* rnnp_b1   = (const float*)d_in[14];
    float* out = (float*)d_out;

    // Workspace carve-out
    char* wsb = (char*)d_ws;
    size_t off = 0;
    auto take = [&](size_t bytes) -> char* {
        char* p = wsb + off;
        off += (bytes + 255) & ~(size_t)255;
        return p;
    };
    int*    spk  = (int*)take((size_t)Bn * Tn * 4);
    int*    pos  = (int*)take((size_t)Bn * Tn * 4);
    __bf16* wRi0  = (__bf16*)take((size_t)2 * G4H * Dn * 2);
    __bf16* wRh0  = (__bf16*)take((size_t)2 * G4H * Hn * 2);
    __bf16* wRi1  = (__bf16*)take((size_t)2 * G4H * H2 * 2);
    __bf16* wRh1  = (__bf16*)take((size_t)2 * G4H * Hn * 2);
    __bf16* wPi0  = (__bf16*)take((size_t)2 * G4H * Dn * 2);
    __bf16* wPh0  = (__bf16*)take((size_t)2 * G4H * Hn * 2);
    __bf16* wPi1  = (__bf16*)take((size_t)2 * G4H * H2 * 2);
    __bf16* wPh1  = (__bf16*)take((size_t)2 * G4H * Hn * 2);
    __bf16* x_in  = (__bf16*)take((size_t)Tn * PBn * Dn * 2);
    __bf16* midb  = (__bf16*)take((size_t)Tn * PBn * H2 * 2);
    __bf16* xg_f  = (__bf16*)take((size_t)Tn * PBn * G4H * 2);
    __bf16* xg_b  = (__bf16*)take((size_t)Tn * PBn * G4H * 2);
    float*  Ebuf  = (float*)take((size_t)Tn * PBn * H2 * 4);
    __bf16* hA    = (__bf16*)take((size_t)2 * PBn * Hn * 2);
    __bf16* hB    = (__bf16*)take((size_t)2 * PBn * Hn * 2);
    float*  cS    = (float*)take((size_t)2 * PBn * Hn * 4);

    auto conv = [&](const float* s, __bf16* d, long long n) {
        k_f32_to_bf16<<<2048, 256, 0, stream>>>(s, d, n);
    };
    auto gemm = [&](const __bf16* X, const __bf16* W, const float* bias,
                    __bf16* Y, int M, int K) {
        long long tiles = (long long)(M / 16) * (G4H / 64);
        k_gemm_xg<<<(int)(tiles / 8), 256, 0, stream>>>(X, W, bias, Y, M, K);
    };
    auto scan = [&](const __bf16* Whh, int NB, __bf16* outb, float* outf,
                    long long ots, int ors) {
        k_zero_u32<<<256, 256, 0, stream>>>((uint32_t*)hA, (long long)(2 * NB * Hn) / 2);
        k_zero_u32<<<256, 256, 0, stream>>>((uint32_t*)cS, (long long)(2 * NB * Hn));
        int grid = 2 * (NB / 16) * 8;
        for (int t = 0; t < Tn; ++t) {
            const __bf16* hin = (t & 1) ? hB : hA;
            __bf16* hout = (t & 1) ? hA : hB;
            k_lstm_step<<<grid, 256, 0, stream>>>(xg_f, xg_b, Whh, hin, hout, cS,
                                                  outb, outf, ots, ors, t, NB);
        }
    };

    // Speaker ranks + weight conversion to bf16 (weights then live in L2: ~50MB < 192MB)
    k_pos<<<1, 64, 0, stream>>>(qmask, spk, pos);
    conv(rnn_Wih0,  wRi0, (long long)2 * G4H * Dn);
    conv(rnn_Whh0,  wRh0, (long long)2 * G4H * Hn);
    conv(rnn_Wih1,  wRi1, (long long)2 * G4H * H2);
    conv(rnn_Whh1,  wRh1, (long long)2 * G4H * Hn);
    conv(rnnp_Wih0, wPi0, (long long)2 * G4H * Dn);
    conv(rnnp_Whh0, wPh0, (long long)2 * G4H * Hn);
    conv(rnnp_Wih1, wPi1, (long long)2 * G4H * H2);
    conv(rnnp_Whh1, wPh1, (long long)2 * G4H * Hn);

    // ---- rnnp (speaker-compacted, P*B = 128 streams) ----
    k_zero_u32<<<4096, 256, 0, stream>>>((uint32_t*)x_in, ((long long)Tn * PBn * Dn) / 2);
    k_gather<<<Tn * Bn, 256, 0, stream>>>(U, spk, pos, x_in);
    {   // layer 0
        int M = Tn * PBn, K = Dn;
        gemm(x_in, wPi0, rnnp_b0, xg_f, M, K);
        gemm(x_in, wPi0 + (size_t)G4H * K, rnnp_b0 + G4H, xg_b, M, K);
        scan(wPh0, PBn, midb, nullptr, 0, 0);
    }
    {   // layer 1 -> E (f32), then scatter into out[:, :, 2H:4H]
        int M = Tn * PBn, K = H2;
        gemm(midb, wPi1, rnnp_b1, xg_f, M, K);
        gemm(midb, wPi1 + (size_t)G4H * K, rnnp_b1 + G4H, xg_b, M, K);
        scan(wPh1, PBn, nullptr, Ebuf, (long long)PBn * H2, H2);
    }
    k_scatter<<<Tn * Bn, 256, 0, stream>>>(Ebuf, spk, pos, out);

    // ---- rnn (raw sequence, B = 64 streams) ----
    conv(U, x_in, (long long)Tn * Bn * Dn);
    {   // layer 0
        int M = Tn * Bn, K = Dn;
        gemm(x_in, wRi0, rnn_b0, xg_f, M, K);
        gemm(x_in, wRi0 + (size_t)G4H * K, rnn_b0 + G4H, xg_b, M, K);
        scan(wRh0, Bn, midb, nullptr, 0, 0);
    }
    {   // layer 1 -> directly into out[:, :, 0:2H] (f32)
        int M = Tn * Bn, K = H2;
        gemm(midb, wRi1, rnn_b1, xg_f, M, K);
        gemm(midb, wRi1 + (size_t)G4H * K, rnn_b1 + G4H, xg_b, M, K);
        scan(wRh1, Bn, nullptr, out, (long long)Bn * 4 * Hn, 4 * Hn);
    }
}
